// GRU_52175262711872
// MI455X (gfx1250) — matrix operand order repaired and checked by reading, not compiled, not run on hardware
//
#include <hip/hip_runtime.h>

// ---------------------------------------------------------------------------
// GRU on MI455X (gfx1250): bf16 WMMA GEMMs + cluster-synchronized recurrence.
//   Phase A: convert x/W/U to bf16 (ws), fused input projection GEMM
//            xproj[t*B+b][3H] = x @ [Wz|Wr|Wh] + bias  (bf16, 192MB ws)
//   Phase B: persistent 16-workgroup scan; WG s owns H-columns [64s,64s+64).
//            step: prefetch next xproj slice; {z,r} = sigmoid(h@U + xproj);
//                  rh = r*h -> L2; cluster barrier;
//                  hcand = tanh(rh@Uh + xproj); h = (1-z)h + z*hcand; barrier.
// ---------------------------------------------------------------------------

typedef __attribute__((ext_vector_type(16))) __bf16 v16bf;
typedef __attribute__((ext_vector_type(8)))  float  v8f;
typedef unsigned short ushort_t;

constexpr int B_ = 64, T_ = 512, I_ = 1024, H_ = 1024;
constexpr int G3H = 3 * H_;

union V16U { v16bf v; uint4 q[2]; };

__device__ __forceinline__ ushort_t f32_to_bf16(float f) {
  union { float f; unsigned u; } v; v.f = f;
  unsigned u = v.u + 0x7FFFu + ((v.u >> 16) & 1u);   // round-to-nearest-even
  return (ushort_t)(u >> 16);
}
__device__ __forceinline__ float bf16_to_f32(ushort_t h) {
  union { unsigned u; float f; } v; v.u = ((unsigned)h) << 16;
  return v.f;
}

// A-tile 16x32 bf16 from row-major [M][K]: lane L<16 -> row L, K=half*8+{0..7},16+half*8+{0..7}
__device__ __forceinline__ v16bf load_a_tile(const ushort_t* __restrict__ base, int lda,
                                             int row_base, int k0, int lane) {
  const int mr = lane & 15, half = lane >> 4;
  const ushort_t* p = base + (size_t)(row_base + mr) * lda + k0 + half * 8;
  V16U r;
  r.q[0] = *(const uint4*)p;
  r.q[1] = *(const uint4*)(p + 16);
  return r.v;
}

// B-tile 32x16 bf16 from row-major [K][N]: lane L -> col n0+L%16, K=half*8+{0..7},16+half*8+{0..7}
__device__ __forceinline__ v16bf load_b_tile(const ushort_t* __restrict__ base, int ldb,
                                             int k0, int n0, int lane) {
  const int half = lane >> 4;
  const ushort_t* p = base + (size_t)(k0 + half * 8) * ldb + n0 + (lane & 15);
  union { v16bf v; ushort_t u[16]; } r;
#pragma unroll
  for (int s = 0; s < 8; ++s) {
    r.u[s]     = p[(size_t)s * ldb];
    r.u[s + 8] = p[(size_t)(16 + s) * ldb];
  }
  return r.v;
}

__device__ __forceinline__ v8f wmma_bf16(v16bf a, v16bf b, v8f c) {
  return __builtin_amdgcn_wmma_f32_16x16x32_bf16(false, a, false, b, (short)0, c, false, false);
}

// Grid-wide sync: hardware cluster barrier when dispatched as a cluster,
// monotonic ticket barrier (agent-scope atomics + s_sleep) otherwise.
__device__ __forceinline__ void grid_sync(unsigned* cnt, unsigned nwg) {
  __syncthreads();
  __threadfence();
  if (__builtin_amdgcn_cluster_id_x() != 0) {
    __builtin_amdgcn_s_cluster_barrier();    // s_barrier_signal -3 ; s_barrier_wait -3
  } else {
    if (threadIdx.x == 0) {
      unsigned ticket = __hip_atomic_fetch_add(cnt, 1u, __ATOMIC_ACQ_REL,
                                               __HIP_MEMORY_SCOPE_AGENT);
      unsigned target = (ticket / nwg + 1u) * nwg;
      while (__hip_atomic_load(cnt, __ATOMIC_ACQUIRE, __HIP_MEMORY_SCOPE_AGENT) < target)
        __builtin_amdgcn_s_sleep(8);
    }
    __syncthreads();
  }
  __threadfence();
}

// --------------------------- prep kernels ----------------------------------

__global__ void cvt_f32_to_bf16(const float* __restrict__ in, ushort_t* __restrict__ out, int n) {
  int i = blockIdx.x * blockDim.x + threadIdx.x;
  int stride = gridDim.x * blockDim.x;
  for (; i < n; i += stride) out[i] = f32_to_bf16(in[i]);
}

__global__ void init_state(ushort_t* __restrict__ h, unsigned* __restrict__ cnt) {
  int i = blockIdx.x * blockDim.x + threadIdx.x;
  if (i < B_ * H_) h[i] = 0;      // bf16 +0.0
  if (i < 64) cnt[i] = 0u;
}

// ------------------- phase A: fused input-projection GEMM ------------------
// grid = (12, 512): WG block = 64 rows x 256 cols of [3H].
// 8 waves/WG, each wave: 2 M-tiles x 4 N-tiles (4 b128 A loads + 64 u16 B loads / 8 WMMA).
__global__ void __launch_bounds__(256) xproj_kernel(
    const ushort_t* __restrict__ xb,
    const ushort_t* __restrict__ Wzb, const ushort_t* __restrict__ Wrb,
    const ushort_t* __restrict__ Whb,
    const float* __restrict__ bz, const float* __restrict__ br, const float* __restrict__ bh,
    ushort_t* __restrict__ xproj)
{
  const int nblk = blockIdx.x;              // 0..11  -> col base 256*nblk in [0,3072)
  const int mblk = blockIdx.y;              // 0..511 -> row base 64*mblk in [0,32768)
  const int wave = threadIdx.x >> 5;
  const int lane = threadIdx.x & 31;
  const int mr = lane & 15, half = lane >> 4;
  const int mw = wave & 1;                  // 2 M-pairs
  const int nw = wave >> 1;                 // 4 N-quads
  const int m0   = mblk * 64 + mw * 32;
  const int ncol = nblk * 256 + nw * 64;    // global fused column, strips +0,16,32,48
  const int g    = nblk >> 2;               // gate: 256-col blocks never straddle gates
  const ushort_t* W   = (g == 0) ? Wzb : (g == 1) ? Wrb : Whb;
  const float*    bia = (g == 0) ? bz  : (g == 1) ? br  : bh;
  const int nloc = ncol - g * H_;

  v8f c00 = {}, c01 = {}, c02 = {}, c03 = {};
  v8f c10 = {}, c11 = {}, c12 = {}, c13 = {};
  for (int k0 = 0; k0 < I_; k0 += 32) {
    v16bf b0 = load_b_tile(W, H_, k0, nloc,      lane);
    v16bf b1 = load_b_tile(W, H_, k0, nloc + 16, lane);
    v16bf b2 = load_b_tile(W, H_, k0, nloc + 32, lane);
    v16bf b3 = load_b_tile(W, H_, k0, nloc + 48, lane);
    v16bf x0 = load_a_tile(xb, I_, m0,      k0, lane);
    v16bf x1 = load_a_tile(xb, I_, m0 + 16, k0, lane);
    c00 = wmma_bf16(x0, b0, c00);
    c01 = wmma_bf16(x0, b1, c01);
    c02 = wmma_bf16(x0, b2, c02);
    c03 = wmma_bf16(x0, b3, c03);
    c10 = wmma_bf16(x1, b0, c10);
    c11 = wmma_bf16(x1, b1, c11);
    c12 = wmma_bf16(x1, b2, c12);
    c13 = wmma_bf16(x1, b3, c13);
  }
  // store: xproj row = t*B + b, where A-row m = b*T + t (time-major for the scan)
  auto store_tile = [&](v8f acc, int mtb, int ntb) {
    const int col = ntb + mr;
    const float bv = bia[col - g * H_];
#pragma unroll
    for (int r2 = 0; r2 < 8; ++r2) {
      int m = mtb + r2 + 8 * half;
      int b_ = m >> 9;                 // m / T
      int t_ = m & (T_ - 1);           // m % T
      xproj[((size_t)t_ * B_ + b_) * G3H + col] = f32_to_bf16(acc[r2] + bv);
    }
  };
  store_tile(c00, m0,      ncol);
  store_tile(c01, m0,      ncol + 16);
  store_tile(c02, m0,      ncol + 32);
  store_tile(c03, m0,      ncol + 48);
  store_tile(c10, m0 + 16, ncol);
  store_tile(c11, m0 + 16, ncol + 16);
  store_tile(c12, m0 + 16, ncol + 32);
  store_tile(c13, m0 + 16, ncol + 48);
}

// ------------------- phase B: persistent recurrent scan --------------------
// 16 WGs x 256 threads; WG s owns H columns [64s, 64s+64).
__global__ void __launch_bounds__(256) gru_scan_kernel(
    const ushort_t* __restrict__ xproj,
    const ushort_t* __restrict__ Uzb, const ushort_t* __restrict__ Urb,
    const ushort_t* __restrict__ Uhb,
    ushort_t* __restrict__ h, ushort_t* __restrict__ rh,
    float* __restrict__ out, unsigned* __restrict__ cnt, unsigned nwg)
{
  __shared__ float zbuf[B_ * 64];           // z gate slice, phase1 -> phase2 (16 KB)
  const int s    = blockIdx.x;              // slice 0..15
  const int wave = threadIdx.x >> 5;
  const int lane = threadIdx.x & 31;
  const int mr   = lane & 15, half = lane >> 4;

  for (int t = 0; t < T_; ++t) {
    // Warm L2 with next step's xproj slice (192MB stream, used exactly once):
    // overlaps the HBM fetch with this step's GEMMs.  -> global_prefetch_b8
    if (t + 1 < T_) {
      const ushort_t* nxt = xproj + (size_t)(t + 1) * B_ * G3H + s * 64;
      const int pb = threadIdx.x >> 2;            // batch row 0..63
      const int pc = (threadIdx.x & 3) * 16;      // 16-col quadrant
#pragma unroll
      for (int g2 = 0; g2 < 3; ++g2)
        __builtin_prefetch(nxt + (size_t)pb * G3H + g2 * H_ + pc, 0, 0);
    }

    // ---- phase 1: waves 0-3 compute z, waves 4-7 compute r (and r*h) ----
    // each wave: 2 M-tiles x 2 N-strips (4 b128 A loads + 32 u16 B loads / 4 WMMA)
    {
      const int g     = wave >> 2;                 // 0:z 1:r
      const int mpair = (wave >> 1) & 1;           // M rows {0,32}
      const int npair = wave & 1;                  // N cols {0,32} within slice
      const int m0    = mpair * 32;
      const int nc0   = s * 64 + npair * 32;       // global H column base
      const ushort_t* U = g ? Urb : Uzb;
      v8f c00 = {}, c01 = {}, c10 = {}, c11 = {};
      for (int k0 = 0; k0 < H_; k0 += 32) {
        v16bf b0 = load_b_tile(U, H_, k0, nc0,      lane);
        v16bf b1 = load_b_tile(U, H_, k0, nc0 + 16, lane);
        v16bf x0 = load_a_tile(h, H_, m0,      k0, lane);
        v16bf x1 = load_a_tile(h, H_, m0 + 16, k0, lane);
        c00 = wmma_bf16(x0, b0, c00);
        c01 = wmma_bf16(x0, b1, c01);
        c10 = wmma_bf16(x1, b0, c10);
        c11 = wmma_bf16(x1, b1, c11);
      }
      auto epi = [&](v8f acc, int mbase, int nct) {
        const int col = nct + mr;
        const ushort_t* xp = xproj + (size_t)t * B_ * G3H + (size_t)g * H_ + col;
#pragma unroll
        for (int r2 = 0; r2 < 8; ++r2) {
          int m = mbase + r2 + 8 * half;
          float pre  = acc[r2] + bf16_to_f32(xp[(size_t)m * G3H]);
          float gate = 1.0f / (1.0f + __expf(-pre));
          if (g == 0) {
            zbuf[m * 64 + (col - s * 64)] = gate;
          } else {
            float hv = bf16_to_f32(h[(size_t)m * H_ + col]);
            rh[(size_t)m * H_ + col] = f32_to_bf16(gate * hv);
          }
        }
      };
      epi(c00, m0,      nc0);
      epi(c01, m0,      nc0 + 16);
      epi(c10, m0 + 16, nc0);
      epi(c11, m0 + 16, nc0 + 16);
    }
    grid_sync(cnt, nwg);    // rh fully published (L2) before anyone consumes it

    // ---- phase 2: hcand = tanh(xh + (r*h)@Uh); h = (1-z)h + z*hcand ----
    {
      const int nstrip = wave & 3;
      const int mpair  = wave >> 2;                // 0..1 -> M rows {0,32}
      const int ncol   = s * 64 + nstrip * 16;
      v8f c0 = {}, c1 = {};
      for (int k0 = 0; k0 < H_; k0 += 32) {
        v16bf bb = load_b_tile(Uhb, H_, k0, ncol, lane);
        v16bf x0 = load_a_tile(rh, H_, mpair * 32,      k0, lane);
        v16bf x1 = load_a_tile(rh, H_, mpair * 32 + 16, k0, lane);
        c0 = wmma_bf16(x0, bb, c0);
        c1 = wmma_bf16(x1, bb, c1);
      }
      const int col = ncol + mr;
      const ushort_t* xp = xproj + (size_t)t * B_ * G3H + (size_t)(2 * H_) + col;
      auto epi2 = [&](v8f acc, int mbase) {
#pragma unroll
        for (int r2 = 0; r2 < 8; ++r2) {
          int m = mbase + r2 + 8 * half;
          float pre   = acc[r2] + bf16_to_f32(xp[(size_t)m * G3H]);
          float hcand = tanhf(pre);
          float z     = zbuf[m * 64 + nstrip * 16 + mr];
          float hold  = bf16_to_f32(h[(size_t)m * H_ + col]);
          float hn    = (1.0f - z) * hold + z * hcand;
          h[(size_t)m * H_ + col] = f32_to_bf16(hn);
          if (t == T_ - 1) out[(size_t)m * H_ + col] = hn;
        }
      };
      epi2(c0, mpair * 32);
      epi2(c1, mpair * 32 + 16);
    }
    grid_sync(cnt, nwg);    // h fully published before next step reads it
  }
}

// ------------------------------- launcher ----------------------------------

extern "C" void kernel_launch(void* const* d_in, const int* in_sizes, int n_in,
                              void* d_out, int out_size, void* d_ws, size_t ws_size,
                              hipStream_t stream)
{
  (void)in_sizes; (void)n_in; (void)out_size; (void)ws_size;
  const float* x  = (const float*)d_in[0];
  const float* Wz = (const float*)d_in[1];
  const float* Uz = (const float*)d_in[2];
  const float* bz = (const float*)d_in[3];
  const float* Wr = (const float*)d_in[4];
  const float* Ur = (const float*)d_in[5];
  const float* br = (const float*)d_in[6];
  const float* Wh = (const float*)d_in[7];
  const float* Uh = (const float*)d_in[8];
  const float* bh = (const float*)d_in[9];

  char* ws = (char*)d_ws;
  size_t off = 0;
  auto carve = [&](size_t bytes) -> void* {
    void* p = ws + off;
    off += (bytes + 255) & ~(size_t)255;
    return p;
  };
  ushort_t* xb    = (ushort_t*)carve((size_t)B_ * T_ * I_ * 2);   // 64 MB
  ushort_t* Wzb   = (ushort_t*)carve((size_t)I_ * H_ * 2);
  ushort_t* Wrb   = (ushort_t*)carve((size_t)I_ * H_ * 2);
  ushort_t* Whb   = (ushort_t*)carve((size_t)I_ * H_ * 2);
  ushort_t* Uzb   = (ushort_t*)carve((size_t)H_ * H_ * 2);
  ushort_t* Urb   = (ushort_t*)carve((size_t)H_ * H_ * 2);
  ushort_t* Uhb   = (ushort_t*)carve((size_t)H_ * H_ * 2);
  ushort_t* xproj = (ushort_t*)carve((size_t)T_ * B_ * G3H * 2);  // 192 MB, [t][b][3H]
  ushort_t* hbuf  = (ushort_t*)carve((size_t)B_ * H_ * 2);
  ushort_t* rhbuf = (ushort_t*)carve((size_t)B_ * H_ * 2);
  unsigned* cnt   = (unsigned*)carve(256);

  // f32 -> bf16 conversions
  cvt_f32_to_bf16<<<4096, 256, 0, stream>>>(x,  xb,  B_ * T_ * I_);
  cvt_f32_to_bf16<<<1024, 256, 0, stream>>>(Wz, Wzb, I_ * H_);
  cvt_f32_to_bf16<<<1024, 256, 0, stream>>>(Wr, Wrb, I_ * H_);
  cvt_f32_to_bf16<<<1024, 256, 0, stream>>>(Wh, Whb, I_ * H_);
  cvt_f32_to_bf16<<<1024, 256, 0, stream>>>(Uz, Uzb, H_ * H_);
  cvt_f32_to_bf16<<<1024, 256, 0, stream>>>(Ur, Urb, H_ * H_);
  cvt_f32_to_bf16<<<1024, 256, 0, stream>>>(Uh, Uhb, H_ * H_);
  init_state<<<(B_ * H_ + 255) / 256, 256, 0, stream>>>(hbuf, cnt);

  // Phase A: full-device WMMA GEMM for input projections (+bias), time-major out
  xproj_kernel<<<dim3(12, 512), 256, 0, stream>>>(xb, Wzb, Wrb, Whb, bz, br, bh, xproj);

  // Phase B: 16-WG persistent scan (cluster-barrier fast path / ticket fallback)
  gru_scan_kernel<<<16, 256, 0, stream>>>(xproj, Uzb, Urb, Uhb, hbuf, rhbuf,
                                          (float*)d_out, cnt, 16u);
}
